// LSTMCell_27762668601814
// MI455X (gfx1250) — compile-verified, matrix operand order checked by reading, not run on hardware
//
#include <hip/hip_runtime.h>
#include <math.h>

typedef _Float16 v16h __attribute__((ext_vector_type(16)));
typedef _Float16 v8h  __attribute__((ext_vector_type(8)));
typedef float    v8f  __attribute__((ext_vector_type(8)));
typedef int      v4i_ __attribute__((ext_vector_type(4)));

#define B_ROWS 8192
#define GEMM_K 1024
#define GEMM_N 4096
#define HDIM   1024

#define BM 256
#define BN 128
#define BK 32
#define NT (GEMM_K / BK)      // 32 K-steps
#define LDS_STR 40            // halves per row: 80B, keeps 16B alignment, spreads banks
#define ASYNC_PER_BATCH 6     // async b128 instructions issued per wave per tile batch

// ---------------- CDNA5 async global->LDS path (guarded) ----------------
#if defined(__has_builtin)
#if __has_builtin(__builtin_amdgcn_global_load_async_to_lds_b128)
#define HAVE_ASYNC 1
#endif
#endif
#ifndef HAVE_ASYNC
#define HAVE_ASYNC 0
#endif

__device__ __forceinline__ void async_copy_b128(const _Float16* g, _Float16* l) {
#if HAVE_ASYNC
    __builtin_amdgcn_global_load_async_to_lds_b128(
        (__attribute__((address_space(1))) v4i_*)g,
        (__attribute__((address_space(3))) v4i_*)l, 0, 0);
#else
    *(uint4*)l = *(const uint4*)g;   // synchronous fallback
#endif
}

template <int N>
__device__ __forceinline__ void wait_async() {
#if HAVE_ASYNC
#if __has_builtin(__builtin_amdgcn_s_wait_asynccnt)
    __builtin_amdgcn_s_wait_asynccnt(N);
#else
    asm volatile("s_wait_asynccnt %0" ::"i"(N) : "memory");
#endif
#endif
}

// ---------------- fp32 -> f16 pre-conversion kernels ----------------
__global__ __launch_bounds__(256) void convert_f16_kernel(const float* __restrict__ src,
                                                          _Float16* __restrict__ dst) {
    int i = (blockIdx.x * 256 + threadIdx.x) * 8;
    float4 a = *(const float4*)(src + i);
    float4 b = *(const float4*)(src + i + 4);
    v8h o;
    o[0] = (_Float16)a.x; o[1] = (_Float16)a.y; o[2] = (_Float16)a.z; o[3] = (_Float16)a.w;
    o[4] = (_Float16)b.x; o[5] = (_Float16)b.y; o[6] = (_Float16)b.z; o[7] = (_Float16)b.w;
    *(v8h*)(dst + i) = o;
}

// W[K][N] fp32 -> WT[N][K] f16 (tiled transpose)
__global__ __launch_bounds__(256) void transpose_convert_f16(const float* __restrict__ W,
                                                             _Float16* __restrict__ WT) {
    __shared__ float tile[32][33];
    const int n0 = blockIdx.x * 32;
    const int k0 = blockIdx.y * 32;
    const int tx = threadIdx.x & 31;
    const int ty = threadIdx.x >> 5;     // 0..7
#pragma unroll
    for (int p = 0; p < 4; ++p) {
        int k = ty + p * 8;
        tile[k][tx] = W[(size_t)(k0 + k) * GEMM_N + n0 + tx];
    }
    __syncthreads();
#pragma unroll
    for (int p = 0; p < 4; ++p) {
        int n = ty + p * 8;
        WT[(size_t)(n0 + n) * GEMM_K + k0 + tx] = (_Float16)tile[tx][n];
    }
}

// ---------------- WMMA fragment load from LDS ----------------
// 16-bit A/B operand per ISA 7.12.2: lane r = row/col (lane&15), kb = 8*(lane>>4);
// halves 0..7 = K {kb..kb+7}, halves 8..15 = K {16+kb..16+kb+7}.
__device__ __forceinline__ v16h load_frag(const _Float16* rowp, int kb) {
    v8h lo = *(const v8h*)(rowp + kb);
    v8h hi = *(const v8h*)(rowp + 16 + kb);
    v16h r;
#pragma unroll
    for (int i = 0; i < 8; ++i) { r[i] = lo[i]; r[8 + i] = hi[i]; }
    return r;
}

// ---------------- GEMM: Y[M,N] = A16[M,K] @ Bt16[N,K]^T ----------------
// 256x128 block tile, 8 waves, 64x64 (4x4 WMMA frags) per wave,
// double-buffered LDS filled by async global->LDS copies.
__global__ __launch_bounds__(256) void gemm_f16_wmma(const _Float16* __restrict__ A,
                                                     const _Float16* __restrict__ Bt,
                                                     float* __restrict__ Y) {
    __shared__ __align__(16) _Float16 As[2][BM * LDS_STR];   // 2 x 20KB
    __shared__ __align__(16) _Float16 Bs[2][BN * LDS_STR];   // 2 x 10KB

    const int t    = threadIdx.x;
    const int lane = t & 31;
    const int wave = t >> 5;
    const int wm   = wave & 3;        // 4 m-groups of 64 rows
    const int wn   = wave >> 2;       // 2 n-groups of 64 cols
    const int r    = lane & 15;
    const int kb   = (lane >> 4) * 8;

    const int m0 = blockIdx.y * BM;
    const int n0 = blockIdx.x * BN;

    const int srow = t >> 2;          // staging: 0..63
    const int sseg = (t & 3) * 8;     // half-offset 0,8,16,24 (16B segments)

    v8f acc[4][4];
#pragma unroll
    for (int i = 0; i < 4; ++i)
#pragma unroll
        for (int j = 0; j < 4; ++j) {
            v8f z = {0.f, 0.f, 0.f, 0.f, 0.f, 0.f, 0.f, 0.f};
            acc[i][j] = z;
        }

    auto issue = [&](int kt, int buf) {
        const int k0 = kt * BK;
#pragma unroll
        for (int p = 0; p < 4; ++p) {           // A: 256 rows x 64B
            int row = srow + p * 64;
            async_copy_b128(A + (size_t)(m0 + row) * GEMM_K + k0 + sseg,
                            &As[buf][row * LDS_STR + sseg]);
        }
#pragma unroll
        for (int p = 0; p < 2; ++p) {           // B: 128 rows x 64B
            int row = srow + p * 64;
            async_copy_b128(Bt + (size_t)(n0 + row) * GEMM_K + k0 + sseg,
                            &Bs[buf][row * LDS_STR + sseg]);
        }
    };

    issue(0, 0);
    issue(1, 1);

    for (int kt = 0; kt < NT; ++kt) {
        const int buf = kt & 1;
        // In-order async completion: one newer batch (6 instrs) may stay in flight.
        if (kt < NT - 1) wait_async<ASYNC_PER_BATCH>();
        else             wait_async<0>();
        __syncthreads();                        // LDS buf visible to all waves

        const _Float16* Ab = &As[buf][0];
        const _Float16* Bb = &Bs[buf][0];
        v16h af[4], bf[4];
#pragma unroll
        for (int i = 0; i < 4; ++i)
            af[i] = load_frag(Ab + (wm * 64 + i * 16 + r) * LDS_STR, kb);
#pragma unroll
        for (int j = 0; j < 4; ++j)
            bf[j] = load_frag(Bb + (wn * 64 + j * 16 + r) * LDS_STR, kb);

#pragma unroll
        for (int i = 0; i < 4; ++i)
#pragma unroll
            for (int j = 0; j < 4; ++j)
                acc[i][j] = __builtin_amdgcn_wmma_f32_16x16x32_f16(
                    false, af[i], false, bf[j], (short)0, acc[i][j], false, false);

        __syncthreads();                        // everyone done reading buf
        if (kt + 2 < NT) issue(kt + 2, buf);    // refill behind compute
    }

    // C/D layout: lane col = lane&15; VGPR v holds row v + 8*(lane>>4)
    const int cn = lane & 15;
    const int ch = lane >> 4;
#pragma unroll
    for (int i = 0; i < 4; ++i)
#pragma unroll
        for (int j = 0; j < 4; ++j)
#pragma unroll
            for (int v = 0; v < 8; ++v) {
                int row = m0 + wm * 64 + i * 16 + v + 8 * ch;
                int col = n0 + wn * 64 + j * 16 + cn;
                Y[(size_t)row * GEMM_N + col] = acc[i][j][v];
            }
}

// ---------------- fused LN1+LN2+gates+cell+LN3 ----------------
__device__ __forceinline__ float sigmoidf_(float x) { return 1.0f / (1.0f + expf(-x)); }

__device__ __forceinline__ float4 block_reduce4(float4 v, float4* red, int t) {
    red[t] = v;
    __syncthreads();
    for (int s = 128; s > 0; s >>= 1) {
        if (t < s) {
            float4 o = red[t + s];
            v.x += o.x; v.y += o.y; v.z += o.z; v.w += o.w;
            red[t] = v;
        }
        __syncthreads();
    }
    float4 out = red[0];
    __syncthreads();
    return out;
}

__global__ __launch_bounds__(256) void lstm_ln_kernel(
    const float* __restrict__ Y1, const float* __restrict__ Y2,
    const float* __restrict__ c0, const float* __restrict__ bias,
    const float* __restrict__ g1, const float* __restrict__ b1,
    const float* __restrict__ g2, const float* __restrict__ b2,
    const float* __restrict__ g3, const float* __restrict__ b3,
    float* __restrict__ h1_out, float* __restrict__ c1_out) {
    const int row = blockIdx.x;
    const int t   = threadIdx.x;
    const float EPS = 1e-5f;

    __shared__ float  gsh[GEMM_N];
    __shared__ float  csh[HDIM];
    __shared__ float4 red[256];

    const float* y1 = Y1 + (size_t)row * GEMM_N;
    const float* y2 = Y2 + (size_t)row * GEMM_N;

    float a[16], b[16];
    float4 st = {0.f, 0.f, 0.f, 0.f};
#pragma unroll
    for (int i = 0; i < 16; ++i) {
        float v = y1[t + 256 * i];
        a[i] = v; st.x += v; st.y += v * v;
    }
#pragma unroll
    for (int i = 0; i < 16; ++i) {
        float v = y2[t + 256 * i];
        b[i] = v; st.z += v; st.w += v * v;
    }
    st = block_reduce4(st, red, t);
    const float inv_n = 1.0f / (float)GEMM_N;
    float mu1 = st.x * inv_n, var1 = st.y * inv_n - mu1 * mu1;
    float mu2 = st.z * inv_n, var2 = st.w * inv_n - mu2 * mu2;
    float rs1 = rsqrtf(var1 + EPS);
    float rs2 = rsqrtf(var2 + EPS);

#pragma unroll
    for (int i = 0; i < 16; ++i) {
        int j = t + 256 * i;
        float g = (a[i] - mu1) * rs1 * g1[j] + b1[j]
                + (b[i] - mu2) * rs2 * g2[j] + b2[j] + bias[j];
        gsh[j] = g;
    }
    __syncthreads();

    float4 st3 = {0.f, 0.f, 0.f, 0.f};
#pragma unroll
    for (int i = 0; i < 4; ++i) {
        int h   = t + 256 * i;
        float fg = gsh[h];
        float ig = gsh[HDIM + h];
        float cg = gsh[3 * HDIM + h];
        float c1 = sigmoidf_(fg) * c0[(size_t)row * HDIM + h]
                 + sigmoidf_(ig) * tanhf(cg);
        csh[h] = c1;
        c1_out[(size_t)row * HDIM + h] = c1;
        st3.x += c1; st3.y += c1 * c1;
    }
    st3 = block_reduce4(st3, red, t);
    const float inv_h = 1.0f / (float)HDIM;
    float mu3 = st3.x * inv_h, var3 = st3.y * inv_h - mu3 * mu3;
    float rs3 = rsqrtf(var3 + EPS);

#pragma unroll
    for (int i = 0; i < 4; ++i) {
        int h   = t + 256 * i;
        float og = gsh[2 * HDIM + h];
        float ln = (csh[h] - mu3) * rs3 * g3[h] + b3[h];
        h1_out[(size_t)row * HDIM + h] = sigmoidf_(og) * tanhf(ln);
    }
}

// ---------------- launch ----------------
extern "C" void kernel_launch(void* const* d_in, const int* in_sizes, int n_in,
                              void* d_out, int out_size, void* d_ws, size_t ws_size,
                              hipStream_t stream) {
    const float* x    = (const float*)d_in[0];
    const float* h0   = (const float*)d_in[1];
    const float* c0   = (const float*)d_in[2];
    const float* Wh   = (const float*)d_in[3];
    const float* Wx   = (const float*)d_in[4];
    const float* bias = (const float*)d_in[5];
    const float* g1   = (const float*)d_in[6];
    const float* b1   = (const float*)d_in[7];
    const float* g2   = (const float*)d_in[8];
    const float* b2   = (const float*)d_in[9];
    const float* g3   = (const float*)d_in[10];
    const float* b3   = (const float*)d_in[11];

    float* h1 = (float*)d_out;
    float* c1 = h1 + (size_t)B_ROWS * HDIM;

    // workspace layout
    float*    Y1   = (float*)d_ws;                              // 128MB
    float*    Y2   = Y1 + (size_t)B_ROWS * GEMM_N;              // 128MB
    _Float16* h016 = (_Float16*)(Y2 + (size_t)B_ROWS * GEMM_N); // 16MB
    _Float16* x16  = h016 + (size_t)B_ROWS * GEMM_K;            // 16MB
    _Float16* WhT  = x16 + (size_t)B_ROWS * GEMM_K;             // 8MB
    _Float16* WxT  = WhT + (size_t)GEMM_N * GEMM_K;             // 8MB

    const int cvt_blocks = (B_ROWS * GEMM_K) / (256 * 8);       // 4096
    convert_f16_kernel<<<cvt_blocks, 256, 0, stream>>>(h0, h016);
    convert_f16_kernel<<<cvt_blocks, 256, 0, stream>>>(x, x16);
    dim3 tgrid(GEMM_N / 32, GEMM_K / 32);                       // (128, 32)
    transpose_convert_f16<<<tgrid, 256, 0, stream>>>(Wh, WhT);
    transpose_convert_f16<<<tgrid, 256, 0, stream>>>(Wx, WxT);

    dim3 ggrid(GEMM_N / BN, B_ROWS / BM);                       // (32, 32)
    gemm_f16_wmma<<<ggrid, 256, 0, stream>>>(h016, WhT, Y1);
    gemm_f16_wmma<<<ggrid, 256, 0, stream>>>(x16,  WxT, Y2);

    lstm_ln_kernel<<<B_ROWS, 256, 0, stream>>>(Y1, Y2, c0, bias,
                                               g1, b1, g2, b2, g3, b3, h1, c1);
}
